// _RPN_66752381714839
// MI455X (gfx1250) — compile-verified, matrix operand order checked by reading, not run on hardware
//
#include <hip/hip_runtime.h>
#include <hip/hip_bf16.h>

typedef __bf16 bf16;
typedef __attribute__((ext_vector_type(8)))  __bf16 bf16x8;
typedef __attribute__((ext_vector_type(16))) __bf16 v16bf;
typedef __attribute__((ext_vector_type(8)))  float  v8f;

// ---------------- problem constants ----------------
#define BB        16
#define CIN       512
#define HH        64
#define HPAD      66            // 64 + 2 halo (zero-padded)
#define PIMG      (HPAD*HPAD)   // 4356 elements per padded channel plane
#define COUT      1024
#define KC        4608          // 9*512
#define NFULL     65536         // B*H*W, ordered ((b,hp,wp) quad, q)
#define NP        16384         // pooled spatial incl. batch
#define AA        9216          // anchors
#define MOFF      378
#define MHAND     27
#define LDSB      40            // LDS row stride (halves): 80B, 16B-aligned chunks

// output layout (floats), concatenated in reference return order
#define OFF_KL    0
#define OFF_CLS   1
#define OFF_HAND  2
#define OFF_LP    (OFF_HAND + (size_t)BB*AA*3)     // 442370
#define OFF_LS    (OFF_LP + BB*42)                 // 443042
#define OFF_LV    (OFF_LS + BB)                    // 443058
#define OFF_RP    (OFF_LV + BB*42)                 // 443730
#define OFF_RS    (OFF_RP + BB*42)                 // 444402
#define OFF_RV    (OFF_RS + BB)                    // 444418

__device__ inline v16bf load_frag(const bf16* row, int half) {
  // 16-bit A/B matrix fragment: two contiguous 16B LDS chunks per lane
  bf16x8 lo = *(const bf16x8*)(row + half * 8);
  bf16x8 hi = *(const bf16x8*)(row + 16 + half * 8);
  return __builtin_shufflevector(lo, hi, 0,1,2,3,4,5,6,7,8,9,10,11,12,13,14,15);
}

__device__ inline v8f zero_v8f() {
  v8f z;
#pragma unroll
  for (int i = 0; i < 8; ++i) z[i] = 0.f;
  return z;
}

__device__ inline bf16x8 zero_bf8() {
  bf16x8 z;
#pragma unroll
  for (int i = 0; i < 8; ++i) z[i] = (bf16)0.0f;
  return z;
}

// ---------------- casts / weight rearrange ----------------
__global__ void cast_f32_bf16(const float* __restrict__ in, bf16* __restrict__ out, int n) {
  for (int i = blockIdx.x * blockDim.x + threadIdx.x; i < n; i += gridDim.x * blockDim.x)
    out[i] = (bf16)in[i];
}

// base_feat (16,512,64,64) f32 -> zero-padded (16,512,66,66) bf16
__global__ void pad_cast_kernel(const float* __restrict__ in, bf16* __restrict__ out) {
  const int total = BB * CIN * PIMG;
  for (int i = blockIdx.x * blockDim.x + threadIdx.x; i < total; i += gridDim.x * blockDim.x) {
    int ww = i % HPAD;
    int t1 = i / HPAD;
    int hh = t1 % HPAD;
    int bc = t1 / HPAD;            // b*512 + c
    int h = hh - 1, w = ww - 1;
    bf16 v = (bf16)0.0f;
    if ((unsigned)h < 64u && (unsigned)w < 64u)
      v = (bf16)in[((size_t)bc * HH + h) * HH + w];
    out[i] = v;
  }
}

// W_conv (1024,512,3,3) -> Wc[m][k], k = tap*512 + c, tap = kh*3+kw
__global__ void rearrange_wconv(const float* __restrict__ in, bf16* __restrict__ out) {
  const int total = COUT * KC;
  for (int i = blockIdx.x * blockDim.x + threadIdx.x; i < total; i += gridDim.x * blockDim.x) {
    int m = i / KC;
    int r = i - m * KC;
    int tap = r >> 9;
    int c   = r & 511;
    out[i] = (bf16)in[((size_t)m * CIN + c) * 9 + tap];
  }
}

// ---------------- conv3x3 (implicit GEMM) + 2x2 avgpool + bias + relu ----------------
// grid: (NFULL/64, COUT/128), block 256 (8 waves). Output x_pooled[p][c] bf16.
// Software-pipelined: chunk k+1 global loads in flight during chunk k WMMAs.
__global__ __launch_bounds__(256)
void conv_pool_kernel(const bf16* __restrict__ Wc, const bf16* __restrict__ Xpad,
                      const float* __restrict__ bias, bf16* __restrict__ Xp) {
  __shared__ bf16 aT[128 * LDSB];
  __shared__ bf16 bT[64 * LDSB];
  const int tid  = threadIdx.x;
  const int lane = tid & 31;
  const int wave = tid >> 5;
  const int nbase = blockIdx.x * 64;
  const int mbase = blockIdx.y * 128;

  // ---- per-thread invariants ----
  // A loader: 128 rows x 32 halves; thread -> (row, 16-half chunk)
  const int arow = tid >> 1, hsel = tid & 1;
  const bf16* asrc = Wc + (size_t)(mbase + arow) * KC + hsel * 16;
  // B loader: thread -> k-pair kp (k0 = 2*kp) and 4 columns n
  const int kp = tid & 15, k0 = kp * 2;
  const int ng = tid >> 4;
  const unsigned short* Xs = (const unsigned short*)Xpad;
  int pixbase[4];
#pragma unroll
  for (int j = 0; j < 4; ++j) {
    int nl = ng * 4 + j;
    int n  = nbase + nl;
    int q  = n & 3;
    int pf = n >> 2;
    int wp = pf & 31, hp = (pf >> 5) & 31, bb = pf >> 10;
    // padded coords: row = hp*2+dy+kh, col = wp*2+dx+kw (kh/kw added per chunk)
    pixbase[j] = bb * (CIN * PIMG) + (hp * 2 + (q >> 1)) * HPAD + (wp * 2 + (q & 1));
  }

  v8f acc[4];
#pragma unroll
  for (int t = 0; t < 4; ++t) acc[t] = zero_v8f();

  bf16x8 pa0, pa1;
  unsigned int pb[4];

  // prefetch helper (branchless: padded input, always-valid addresses)
  auto prefetch = [&](int chunk) {
    pa0 = *(const bf16x8*)(asrc + chunk * 32);
    pa1 = *(const bf16x8*)(asrc + chunk * 32 + 8);
    const int tap = chunk >> 4;
    const int kh = tap / 3, kw = tap - kh * 3;
    const int c  = ((chunk & 15) << 5) + k0;
    const int coff = c * PIMG + kh * HPAD + kw;
#pragma unroll
    for (int j = 0; j < 4; ++j) {
      int addr = pixbase[j] + coff;
      unsigned int lo = Xs[addr];
      unsigned int hi = Xs[addr + PIMG];
      pb[j] = lo | (hi << 16);
    }
  };

  prefetch(0);
  for (int chunk = 0; chunk < 144; ++chunk) {
    __syncthreads();                       // previous chunk's LDS reads complete
    // commit prefetched regs to LDS
    *(bf16x8*)(aT + arow * LDSB + hsel * 16)     = pa0;
    *(bf16x8*)(aT + arow * LDSB + hsel * 16 + 8) = pa1;
#pragma unroll
    for (int j = 0; j < 4; ++j)
      *(unsigned int*)(bT + (ng * 4 + j) * LDSB + k0) = pb[j];
    if (chunk + 1 < 144) prefetch(chunk + 1);  // overlap with compute below
    __syncthreads();                       // LDS tile ready

    const int half = lane >> 4, lm = lane & 15;
    v16bf a = load_frag(aT + (wave * 16 + lm) * LDSB, half);
    v16bf bfr[4];
#pragma unroll
    for (int t = 0; t < 4; ++t)
      bfr[t] = load_frag(bT + (t * 16 + lm) * LDSB, half);
#pragma unroll
    for (int t = 0; t < 4; ++t)
      acc[t] = __builtin_amdgcn_wmma_f32_16x16x32_bf16(false, a, false, bfr[t],
                                                       (short)0, acc[t], false, false);
  }

  // epilogue: 2x2 pool over 4 adjacent lanes (quad columns), +bias, relu, bf16
  const int half = lane >> 4, laneN = lane & 15;
  const int m0 = mbase + wave * 16 + half * 8;
#pragma unroll
  for (int t = 0; t < 4; ++t) {
    float ps[8];
#pragma unroll
    for (int r = 0; r < 8; ++r) {
      float v = acc[t][r];
      v += __shfl_xor(v, 1);
      v += __shfl_xor(v, 2);
      ps[r] = v;
    }
    if ((laneN & 3) == 0) {
      int pg = (nbase + t * 16 + laneN) >> 2;
      bf16x8 o;
#pragma unroll
      for (int r = 0; r < 8; ++r) {
        float v = ps[r] * 0.25f + bias[m0 + r];
        o[r] = (bf16)(v > 0.f ? v : 0.f);
      }
      *(bf16x8*)(Xp + (size_t)pg * COUT + m0) = o;
    }
  }
}

// ---------------- 1x1 head GEMM: Y[m][NP] = Wb[m][1024] @ Xp[p][1024]^T + bias ----------------
// grid: (NP/64, ceil(M/64)), block 256 (8 waves: 4 M x 2 N), software-pipelined
__global__ __launch_bounds__(256)
void head_gemm(const bf16* __restrict__ Wb, const float* __restrict__ bias,
               const bf16* __restrict__ Xp, float* __restrict__ Y, int M) {
  __shared__ bf16 aT[64 * LDSB];
  __shared__ bf16 bT[64 * LDSB];
  const int tid = threadIdx.x, lane = tid & 31, wave = tid >> 5;
  const int my = wave >> 1, nx = wave & 1;
  const int nbase = blockIdx.x * 64;
  const int mbase = blockIdx.y * 64;
  const int row = tid >> 2, ch = tid & 3;
  const int mrow = mbase + row;
  const bool mok = (mrow < M);

  v8f acc[2];
  acc[0] = zero_v8f(); acc[1] = zero_v8f();

  bf16x8 pa, pbv;
  auto prefetch = [&](int kk) {
    pa = zero_bf8();
    if (mok) pa = *(const bf16x8*)(Wb + (size_t)mrow * 1024 + kk * 32 + ch * 8);
    pbv = *(const bf16x8*)(Xp + (size_t)(nbase + row) * 1024 + kk * 32 + ch * 8);
  };

  prefetch(0);
  for (int kk = 0; kk < 32; ++kk) {
    __syncthreads();
    *(bf16x8*)(aT + row * LDSB + ch * 8) = pa;
    *(bf16x8*)(bT + row * LDSB + ch * 8) = pbv;
    if (kk + 1 < 32) prefetch(kk + 1);
    __syncthreads();
    const int half = lane >> 4, lm = lane & 15;
    v16bf a = load_frag(aT + (my * 16 + lm) * LDSB, half);
    v16bf b0 = load_frag(bT + (nx * 32 + lm) * LDSB, half);
    v16bf b1 = load_frag(bT + (nx * 32 + 16 + lm) * LDSB, half);
    acc[0] = __builtin_amdgcn_wmma_f32_16x16x32_bf16(false, a, false, b0,
                                                     (short)0, acc[0], false, false);
    acc[1] = __builtin_amdgcn_wmma_f32_16x16x32_bf16(false, a, false, b1,
                                                     (short)0, acc[1], false, false);
  }
  const int half = lane >> 4, lm = lane & 15;
#pragma unroll
  for (int t = 0; t < 2; ++t) {
    int p = nbase + nx * 32 + t * 16 + lm;
#pragma unroll
    for (int r = 0; r < 8; ++r) {
      int m = mbase + my * 16 + half * 8 + r;
      if (m < M) Y[(size_t)m * NP + p] = acc[t][r] + bias[m];
    }
  }
}

// ---------------- per-batch: softmax/hand, keeps, top-10 negs, cls partial ----------------
__device__ void argmax_reduce(volatile float* sv, volatile int* si, int tid) {
  __syncthreads();
  for (int s = 128; s > 0; s >>= 1) {
    if (tid < s) {
      float v2 = sv[tid + s]; int i2 = si[tid + s];
      if (v2 > sv[tid] || (v2 == sv[tid] && i2 < si[tid])) { sv[tid] = v2; si[tid] = i2; }
    }
    __syncthreads();
  }
}

__global__ __launch_bounds__(256)
void batch_cls_kernel(const float* __restrict__ Yh, const int* __restrict__ inds_inside,
                      const int* __restrict__ labels, const float* __restrict__ neg_rand,
                      float* __restrict__ out, int* __restrict__ sel_i,
                      float* __restrict__ sel_f, float* __restrict__ cls_part) {
  const int b = blockIdx.x, tid = threadIdx.x;
  __shared__ float sv[256];
  __shared__ int   si[256];
  __shared__ int   chosen[10];
  __shared__ int   sLidx, sRidx;
  __shared__ float sLsc, sRsc;
  const float NEG_INF = -1e30f;

  float lbest = NEG_INF, rbest = NEG_INF; int lbi = 0x7fffffff, rbi = 0x7fffffff;
  for (int a = tid; a < AA; a += 256) {
    int a9 = a >> 10, pix = a & 1023;
    int p = b * 1024 + pix;
    float l0 = Yh[(size_t)(a9) * NP + p];
    float l1 = Yh[(size_t)(9 + a9) * NP + p];
    float l2 = Yh[(size_t)(18 + a9) * NP + p];
    float mx = fmaxf(l0, fmaxf(l1, l2));
    float e0 = __expf(l0 - mx), e1 = __expf(l1 - mx), e2 = __expf(l2 - mx);
    float s = e0 + e1 + e2, inv = 1.f / s;
    float h0 = e0 * inv, h1 = e1 * inv, h2 = e2 * inv;
    size_t ho = OFF_HAND + ((size_t)b * AA + a) * 3;
    out[ho] = h0; out[ho + 1] = h1; out[ho + 2] = h2;
    int inside = inds_inside[a];
    bool lp_ = (h0 > h1) && (h0 > h2);
    bool rp_ = (h2 > h0) && (h2 > h1);
    if (lp_ && inside && (h0 > lbest || (h0 == lbest && a < lbi))) { lbest = h0; lbi = a; }
    if (rp_ && inside && (h2 > rbest || (h2 == rbest && a < rbi))) { rbest = h2; rbi = a; }
  }
  sv[tid] = lbest; si[tid] = lbi; argmax_reduce(sv, si, tid);
  if (tid == 0) { sLsc = sv[0]; sLidx = (sv[0] > NEG_INF * 0.5f) ? si[0] : 0; }
  __syncthreads();
  sv[tid] = rbest; si[tid] = rbi; argmax_reduce(sv, si, tid);
  if (tid == 0) { sRsc = sv[0]; sRidx = (sv[0] > NEG_INF * 0.5f) ? si[0] : 0; }
  __syncthreads();

  for (int it = 0; it < 10; ++it) {
    float best = NEG_INF; int bidx = 0x7fffffff;
    for (int a = tid; a < AA; a += 256) {
      if (labels[((size_t)b * AA + a) * 3 + 1] == 1) {
        bool used = false;
        for (int c = 0; c < it; ++c) if (chosen[c] == a) used = true;
        if (!used) {
          float r = neg_rand[(size_t)b * AA + a];
          if (r > best || (r == best && a < bidx)) { best = r; bidx = a; }
        }
      }
    }
    sv[tid] = best; si[tid] = bidx; argmax_reduce(sv, si, tid);
    if (tid == 0) chosen[it] = (sv[0] > NEG_INF * 0.5f) ? si[0] : -1;
    __syncthreads();
  }

  float ce_sum = 0.f; int cnt = 0;
  for (int a = tid; a < AA; a += 256) {
    int a9 = a >> 10, pix = a & 1023;
    int p = b * 1024 + pix;
    float l0 = Yh[(size_t)(a9) * NP + p];
    float l1 = Yh[(size_t)(9 + a9) * NP + p];
    float l2 = Yh[(size_t)(18 + a9) * NP + p];
    float mx = fmaxf(l0, fmaxf(l1, l2));
    float e0 = __expf(l0 - mx), e1 = __expf(l1 - mx), e2 = __expf(l2 - mx);
    float s = e0 + e1 + e2, inv = 1.f / s;
    float h0 = e0 * inv, h1 = e1 * inv, h2 = e2 * inv;
    float lse = mx + __logf(s);
    int lab0 = labels[((size_t)b * AA + a) * 3];
    int lab1 = labels[((size_t)b * AA + a) * 3 + 1];
    int lab2 = labels[((size_t)b * AA + a) * 3 + 2];
    int tcls = 0; int bl = lab0;
    if (lab1 > bl) { tcls = 1; bl = lab1; }
    if (lab2 > bl) { tcls = 2; }
    float lt = (tcls == 0) ? l0 : ((tcls == 1) ? l1 : l2);
    float ce = lse - lt;
    int inside = inds_inside[a];
    bool lk = (h0 > h1) && (h0 > h2) && inside;
    bool rk = (h2 > h0) && (h2 > h1) && inside;
    bool ns = false;
    if (lab1 == 1)
      for (int c = 0; c < 10; ++c) if (chosen[c] == a) ns = true;
    if (lk || rk || ns) { ce_sum += ce; cnt++; }
  }
  sv[tid] = ce_sum; si[tid] = cnt; __syncthreads();
  for (int s = 128; s > 0; s >>= 1) {
    if (tid < s) { sv[tid] += sv[tid + s]; si[tid] += si[tid + s]; }
    __syncthreads();
  }
  if (tid == 0) {
    float c = (float)(si[0] > 1 ? si[0] : 1);
    cls_part[b] = sv[0] / (c * c);
    int lhas = (sLsc > NEG_INF * 0.5f) ? 1 : 0;
    int rhas = (sRsc > NEG_INF * 0.5f) ? 1 : 0;
    sel_i[b * 4 + 0] = sLidx; sel_i[b * 4 + 1] = lhas;
    sel_i[b * 4 + 2] = sRidx; sel_i[b * 4 + 3] = rhas;
    sel_f[b * 2 + 0] = lhas ? sLsc : 0.f;
    sel_f[b * 2 + 1] = rhas ? sRsc : 0.f;
  }
}

// ---------------- per (batch, side): pose/var outputs + KL partial ----------------
__global__ __launch_bounds__(64)
void kl_kernel(const float* __restrict__ Yoff, const float* __restrict__ Yvar,
               const float* __restrict__ anchors, const float* __restrict__ pose_gt,
               const int* __restrict__ hand_type, const int* __restrict__ sel_i,
               const float* __restrict__ sel_f, float* __restrict__ out,
               float* __restrict__ kl_part) {
  const int b = blockIdx.x, side = blockIdx.y, tid = threadIdx.x;
  const int idx = sel_i[b * 4 + side * 2];
  const int has = sel_i[b * 4 + side * 2 + 1];
  const float sc = sel_f[b * 2 + side];
  __shared__ float aWH[2];
  __shared__ float pose[42], varr[42], tgt[42];
  const int a9 = idx >> 10, pix = idx & 1023;
  const int p = b * 1024 + pix;

  if (tid == 0) {
    float mnx = 3.4e38f, mxx = -3.4e38f, mny = 3.4e38f, mxy = -3.4e38f;
    for (int j = 0; j < 21; ++j) {
      float ax = anchors[(size_t)idx * 42 + j * 2];
      float ay = anchors[(size_t)idx * 42 + j * 2 + 1];
      mnx = fminf(mnx, ax); mxx = fmaxf(mxx, ax);
      mny = fminf(mny, ay); mxy = fmaxf(mxy, ay);
    }
    aWH[0] = mxx - mnx; aWH[1] = mxy - mny;
  }
  __syncthreads();
  if (tid < 42) {
    int coord = tid & 1, joint = tid >> 1;
    float anc = anchors[(size_t)idx * 42 + tid];
    float off = Yoff[(size_t)(tid * 9 + a9) * NP + p];
    float vr  = Yvar[(size_t)(tid * 9 + a9) * NP + p];
    float ps = anc + off * aWH[coord];
    float vv = __expf(vr);
    if (!has) { ps = 0.f; vv = 0.f; }
    pose[tid] = ps; varr[tid] = vv;
    int ht = hand_type[b * 2 + (side == 0 ? 1 : 0)];
    int offj = (side == 0) ? ((ht == 1) ? 21 : 0) : ((ht == 1) ? 0 : 21);
    tgt[tid] = pose_gt[((size_t)b * 42 + offj + joint) * 3 + coord];
    size_t pbase = (side == 0) ? OFF_LP : OFF_RP;
    size_t vbase = (side == 0) ? OFF_LV : OFF_RV;
    out[pbase + b * 42 + tid] = ps;
    out[vbase + b * 42 + tid] = vv;
  }
  __syncthreads();
  if (tid == 0) {
    size_t sbase = (side == 0) ? OFF_LS : OFF_RS;
    out[sbase + b] = has ? sc : 0.f;
    float kl = 0.f;
    if (has) {
      float mnx = 3.4e38f, mxx = -3.4e38f, mny = 3.4e38f, mxy = -3.4e38f;
      for (int j = 0; j < 21; ++j) {
        mnx = fminf(mnx, pose[j * 2]);     mxx = fmaxf(mxx, pose[j * 2]);
        mny = fminf(mny, pose[j * 2 + 1]); mxy = fmaxf(mxy, pose[j * 2 + 1]);
      }
      float scale = 2.f - (mxx - mnx) * (mxy - mny) / 65536.f;
      float sum = 0.f;
      for (int t = 0; t < 42; ++t) {
        float d = pose[t] - tgt[t], ad = fabsf(d);
        float sl = (ad < 1.f) ? 0.5f * d * d : ad - 0.5f;
        sum += __expf(-varr[t]) * sl + 0.5f * varr[t];
      }
      kl = sc * scale * sum;  // KL_WEIGHT = 1
    }
    kl_part[b * 2 + side] = kl;
  }
}

__global__ void finalize_kernel(const float* __restrict__ kl_part,
                                const float* __restrict__ cls_part, float* __restrict__ out) {
  if (threadIdx.x == 0 && blockIdx.x == 0) {
    float k = 0.f;
    for (int i = 0; i < 32; ++i) k += kl_part[i];
    float c = 0.f;
    for (int i = 0; i < 16; ++i) c += cls_part[i];
    out[OFF_KL] = k;
    out[OFF_CLS] = c;
  }
}

// ---------------- host launcher ----------------
extern "C" void kernel_launch(void* const* d_in, const int* in_sizes, int n_in,
                              void* d_out, int out_size, void* d_ws, size_t ws_size,
                              hipStream_t stream) {
  (void)in_sizes; (void)n_in; (void)out_size; (void)ws_size;
  const float* base   = (const float*)d_in[0];
  const float* posegt = (const float*)d_in[1];
  const float* anch   = (const float*)d_in[2];
  const float* negr   = (const float*)d_in[3];
  const float* Wconv  = (const float*)d_in[4];
  const float* bconv  = (const float*)d_in[5];
  const float* Woff   = (const float*)d_in[6];
  const float* boff   = (const float*)d_in[7];
  const float* Wvar   = (const float*)d_in[8];
  const float* bvar   = (const float*)d_in[9];
  const float* Whand  = (const float*)d_in[10];
  const float* bhand  = (const float*)d_in[11];
  const int*   htype  = (const int*)d_in[12];
  const int*   inside = (const int*)d_in[13];
  const int*   labels = (const int*)d_in[14];
  float* out = (float*)d_out;

  char* w = (char*)d_ws;
  auto alloc = [&](size_t bytes) -> void* {
    void* r = (void*)w;
    w += (bytes + 255) & ~(size_t)255;
    return r;
  };
  bf16* base_bf = (bf16*)alloc((size_t)BB * CIN * PIMG * 2);          // 71 MB padded
  bf16* Wc_bf   = (bf16*)alloc((size_t)COUT * KC * 2);                // 9.4 MB
  bf16* Woff_bf = (bf16*)alloc((size_t)MOFF * 1024 * 2);
  bf16* Wvar_bf = (bf16*)alloc((size_t)MOFF * 1024 * 2);
  bf16* Whand_bf= (bf16*)alloc((size_t)MHAND * 1024 * 2);
  bf16* Xp      = (bf16*)alloc((size_t)NP * COUT * 2);                // 33.5 MB
  float* Yoff   = (float*)alloc((size_t)MOFF * NP * 4);               // 24.8 MB
  float* Yvar   = (float*)alloc((size_t)MOFF * NP * 4);
  float* Yhand  = (float*)alloc((size_t)MHAND * NP * 4);
  int*   sel_i  = (int*)alloc(BB * 4 * 4);
  float* sel_f  = (float*)alloc(BB * 2 * 4);
  float* clsp   = (float*)alloc(BB * 4);
  float* klp    = (float*)alloc(BB * 2 * 4);

  // 1) casts / padding / weight rearrange
  pad_cast_kernel<<<8192, 256, 0, stream>>>(base, base_bf);
  rearrange_wconv<<<4608, 256, 0, stream>>>(Wconv, Wc_bf);
  cast_f32_bf16<<<1536, 256, 0, stream>>>(Woff, Woff_bf, MOFF * 1024);
  cast_f32_bf16<<<1536, 256, 0, stream>>>(Wvar, Wvar_bf, MOFF * 1024);
  cast_f32_bf16<<<112, 256, 0, stream>>>(Whand, Whand_bf, MHAND * 1024);
  // 2) conv3x3 + pool + relu (WMMA implicit GEMM)
  conv_pool_kernel<<<dim3(NFULL / 64, COUT / 128), 256, 0, stream>>>(Wc_bf, base_bf, bconv, Xp);
  // 3) heads (WMMA GEMM)
  head_gemm<<<dim3(NP / 64, 6), 256, 0, stream>>>(Woff_bf, boff, Xp, Yoff, MOFF);
  head_gemm<<<dim3(NP / 64, 6), 256, 0, stream>>>(Wvar_bf, bvar, Xp, Yvar, MOFF);
  head_gemm<<<dim3(NP / 64, 1), 256, 0, stream>>>(Whand_bf, bhand, Xp, Yhand, MHAND);
  // 4) per-batch classification / selection
  batch_cls_kernel<<<BB, 256, 0, stream>>>(Yhand, inside, labels, negr, out, sel_i, sel_f, clsp);
  // 5) per-(batch,side) KL + gathered outputs
  kl_kernel<<<dim3(BB, 2), 64, 0, stream>>>(Yoff, Yvar, anch, posegt, htype, sel_i, sel_f, out, klp);
  // 6) deterministic reduction of loss partials
  finalize_kernel<<<1, 64, 0, stream>>>(klp, clsp, out);
}
